// PPI_GCN_24429773979884
// MI455X (gfx1250) — compile-verified, matrix operand order checked by reading, not run on hardware
//
#include <hip/hip_runtime.h>
#include <hip/hip_bf16.h>
#include <math.h>

typedef __attribute__((ext_vector_type(16))) _Float16 v16h;
typedef __attribute__((ext_vector_type(8)))  _Float16 v8h;
typedef __attribute__((ext_vector_type(8)))  float    v8f;

#define NN 50000
#define EE 800000
#define BBG 512
#define FFD 33

// ---------------- pad/convert kernels (coalesced streaming, trivial cost) ----------------

// Apad[Mpad x Kpad] (f16, K-major) = pad(A[M x K] f32)
__global__ void cvt_a_pad_kernel(const float* __restrict__ A, _Float16* __restrict__ Apad,
                                 int M, int K, int Kpad, int total) {
  int i = blockIdx.x * blockDim.x + threadIdx.x;
  if (i >= total) return;
  int row = i / Kpad, k = i - row * Kpad;
  float v = (row < M && k < K) ? A[(size_t)row * K + k] : 0.0f;
  Apad[i] = (_Float16)v;
}

// Bt[Npad x Kpad] (f16, K-major) = pad(transpose(W[K x N] f32))
__global__ void cvt_bt_pad_kernel(const float* __restrict__ W, _Float16* __restrict__ Bt,
                                  int K, int N, int Kpad, int total) {
  int i = blockIdx.x * blockDim.x + threadIdx.x;
  if (i >= total) return;
  int n = i / Kpad, k = i - n * Kpad;
  float v = (n < N && k < K) ? W[(size_t)k * N + n] : 0.0f;
  Bt[i] = (_Float16)v;
}

// ---------------- elementwise / scatter kernels ----------------

__global__ void deg_init_kernel(float* __restrict__ deg, int n) {
  int i = blockIdx.x * blockDim.x + threadIdx.x;
  if (i < n) deg[i] = 1.0f;                       // self loop
}

__global__ void deg_count_kernel(const int* __restrict__ dst, float* __restrict__ deg, int e) {
  int i = blockIdx.x * blockDim.x + threadIdx.x;
  if (i < e) atomicAdd(&deg[dst[i]], 1.0f);
}

__global__ void deg_inv_kernel(float* __restrict__ deg, int n) {
  int i = blockIdx.x * blockDim.x + threadIdx.x;
  if (i < n) deg[i] = rsqrtf(deg[i]);             // deg >= 1 always
}

// out[i,c] = h[i,c]*dinv[i]^2 + bias[c]   (self-loop term + bias)
__global__ void agg_init_kernel(const float* __restrict__ h, const float* __restrict__ dinv,
                                const float* __restrict__ bias, float* __restrict__ out,
                                int n, int C) {
  int i = blockIdx.x * blockDim.x + threadIdx.x;
  if (i < n * C) {
    int node = i / C, c = i - node * C;
    float di = dinv[node];
    out[i] = h[i] * di * di + bias[c];
  }
}

// wave per edge; lanes sweep channels (coalesced load of h[src,:], atomic add to out[dst,:])
__global__ void edge_scatter_kernel(const float* __restrict__ h, const int* __restrict__ src,
                                    const int* __restrict__ dst, const float* __restrict__ dinv,
                                    float* __restrict__ out, int e, int C) {
  int wave = (blockIdx.x * blockDim.x + threadIdx.x) >> 5;
  int lane = threadIdx.x & 31;
  if (wave >= e) return;
  int s = src[wave], d = dst[wave];
  float nrm = dinv[s] * dinv[d];
  const float* hs = h + (size_t)s * C;
  float* od = out + (size_t)d * C;
  for (int c = lane; c < C; c += 32)
    atomicAdd(&od[c], hs[c] * nrm);
}

__global__ void relu_kernel(float* __restrict__ x, int n) {
  int i = blockIdx.x * blockDim.x + threadIdx.x;
  if (i < n) x[i] = fmaxf(x[i], 0.0f);
}

__global__ void pool_zero_kernel(float* __restrict__ pool, float* __restrict__ cnt, int nb, int C) {
  int i = blockIdx.x * blockDim.x + threadIdx.x;
  if (i < nb * C) pool[i] = 0.0f;
  if (i < nb) cnt[i] = 0.0f;
}

__global__ void pool_acc_kernel(const float* __restrict__ x, const int* __restrict__ batch,
                                float* __restrict__ pool, float* __restrict__ cnt, int n, int C) {
  int wave = (blockIdx.x * blockDim.x + threadIdx.x) >> 5;
  int lane = threadIdx.x & 31;
  if (wave >= n) return;
  int b = batch[wave];
  const float* xs = x + (size_t)wave * C;
  float* pb = pool + (size_t)b * C;
  for (int c = lane; c < C; c += 32) atomicAdd(&pb[c], xs[c]);
  if (lane == 0) atomicAdd(&cnt[b], 1.0f);
}

__global__ void pool_div_kernel(float* __restrict__ pool, const float* __restrict__ cnt,
                                int nb, int C) {
  int i = blockIdx.x * blockDim.x + threadIdx.x;
  if (i < nb * C) pool[i] /= fmaxf(cnt[i / C], 1.0f);
}

// final 512-dot + sigmoid, wave32 shuffle reduction
__global__ void out_layer_kernel(const float* __restrict__ x, const float* __restrict__ W,
                                 const float* __restrict__ b, float* __restrict__ out,
                                 int rows, int K) {
  int wave = (blockIdx.x * blockDim.x + threadIdx.x) >> 5;
  int lane = threadIdx.x & 31;
  if (wave >= rows) return;
  const float* xr = x + (size_t)wave * K;
  float s = 0.0f;
  for (int k = lane; k < K; k += 32) s += xr[k] * W[k];
  for (int o = 16; o > 0; o >>= 1) s += __shfl_xor(s, o, 32);
  if (lane == 0) out[wave] = 1.0f / (1.0f + expf(-(s + b[0])));
}

// ---------------- WMMA GEMM on pre-padded f16 operands ----------------
// C[M,Nn] = Apad[Mpad,Kpad] @ Bt[Npad,Kpad]^T (+bias)(+relu)
// One wave per 16x16 output tile; fragments are two contiguous 16B chunks per lane.

__device__ __forceinline__ v16h load_frag(const _Float16* __restrict__ base) {
  v8h lo = *(const v8h*)(base);        // K = kb .. kb+7
  v8h hi = *(const v8h*)(base + 16);   // K = kb+16 .. kb+23
  v16h f;
#pragma unroll
  for (int j = 0; j < 8; ++j) { f[j] = lo[j]; f[j + 8] = hi[j]; }
  return f;
}

__global__ void wmma_gemm_kernel(const _Float16* __restrict__ Apad,
                                 const _Float16* __restrict__ Bt,
                                 const float* __restrict__ bias, float* __restrict__ C,
                                 int M, int Nn, int Kpad, int ldc, int coff, int act,
                                 int tilesN, int numTiles) {
  int wave = (blockIdx.x * blockDim.x + threadIdx.x) >> 5;
  if (wave >= numTiles) return;                 // wave-uniform: EXEC all-1s at WMMA
  int lane = threadIdx.x & 31;
  int tm = wave / tilesN, tn = wave - tm * tilesN;
  int rowBase = tm << 4, colBase = tn << 4;
  int halfSel = lane >> 4;
  int l15 = lane & 15;
  int kb = halfSel << 3;                        // lane-half K sub-slab offset

  const _Float16* ap = Apad + (size_t)(rowBase + l15) * Kpad + kb;  // always in padded range
  const _Float16* bp = Bt   + (size_t)(colBase + l15) * Kpad + kb;

  v8f acc0 = {0.f, 0.f, 0.f, 0.f, 0.f, 0.f, 0.f, 0.f};
  v8f acc1 = {0.f, 0.f, 0.f, 0.f, 0.f, 0.f, 0.f, 0.f};

  int k0 = 0;
  for (; k0 + 64 <= Kpad; k0 += 64) {           // 2 accumulators break the WMMA RAW chain
    __builtin_prefetch(ap + k0 + 64, 0, 3);     // global_prefetch_b8 next slab
    v16h a0 = load_frag(ap + k0);
    v16h b0 = load_frag(bp + k0);
    v16h a1 = load_frag(ap + k0 + 32);
    v16h b1 = load_frag(bp + k0 + 32);
    acc0 = __builtin_amdgcn_wmma_f32_16x16x32_f16(false, a0, false, b0, (short)0, acc0,
                                                  false, false);
    acc1 = __builtin_amdgcn_wmma_f32_16x16x32_f16(false, a1, false, b1, (short)0, acc1,
                                                  false, false);
  }
  if (k0 < Kpad) {                              // Kpad is a 32-multiple; one tail slab
    v16h a0 = load_frag(ap + k0);
    v16h b0 = load_frag(bp + k0);
    acc0 = __builtin_amdgcn_wmma_f32_16x16x32_f16(false, a0, false, b0, (short)0, acc0,
                                                  false, false);
  }

  int bcol = colBase + l15;
#pragma unroll
  for (int r = 0; r < 8; ++r) {                 // C/D layout: VGPR r -> M = r + 8*laneHalf
    int row = rowBase + r + (halfSel << 3);
    if (row < M && bcol < Nn) {
      float v = acc0[r] + acc1[r];
      if (bias) v += bias[bcol];
      if (act) v = fmaxf(v, 0.0f);
      C[(size_t)row * ldc + coff + bcol] = v;
    }
  }
}

// ---------------- host-side orchestration ----------------

static inline int cdiv(int a, int b) { return (a + b - 1) / b; }

static void run_gemm(const float* A, const float* Wf32, const float* bias, float* C,
                     int M, int Nn, int K, int ldc, int coff, int act,
                     _Float16* Apad, _Float16* Bt, hipStream_t stream) {
  int Kpad   = cdiv(K, 32) * 32;
  int tilesM = cdiv(M, 16), tilesN = cdiv(Nn, 16);
  int Mpad = tilesM * 16, Npad = tilesN * 16;

  int aTot = Mpad * Kpad;
  cvt_a_pad_kernel<<<cdiv(aTot, 256), 256, 0, stream>>>(A, Apad, M, K, Kpad, aTot);
  int bTot = Npad * Kpad;
  cvt_bt_pad_kernel<<<cdiv(bTot, 256), 256, 0, stream>>>(Wf32, Bt, K, Nn, Kpad, bTot);

  int numTiles = tilesM * tilesN;
  wmma_gemm_kernel<<<cdiv(numTiles, 8), 256, 0, stream>>>(Apad, Bt, bias, C, M, Nn, Kpad,
                                                          ldc, coff, act, tilesN, numTiles);
}

extern "C" void kernel_launch(void* const* d_in, const int* in_sizes, int n_in,
                              void* d_out, int out_size, void* d_ws, size_t ws_size,
                              hipStream_t stream) {
  (void)in_sizes; (void)n_in; (void)out_size; (void)ws_size;

  char* w = (char*)d_ws;
  size_t off = 0;
  auto carve = [&](size_t bytes) -> void* {
    void* p = w + off;
    off += (bytes + 255) & ~(size_t)255;
    return p;
  };
  float*    bufH = (float*)carve((size_t)NN * 132 * 4);     // GEMM output h
  float*    bufX = (float*)carve((size_t)NN * 132 * 4);     // aggregation output / next input
  float*    dinv = (float*)carve((size_t)NN * 4);
  float*    pool = (float*)carve((size_t)BBG * 132 * 4);
  float*    cnt  = (float*)carve((size_t)BBG * 4);
  float*    g1   = (float*)carve((size_t)512 * 1024 * 4);
  float*    xc   = (float*)carve((size_t)512 * 256 * 4);
  float*    f1   = (float*)carve((size_t)512 * 1024 * 4);
  float*    f2   = (float*)carve((size_t)512 * 512 * 4);
  _Float16* Apad = (_Float16*)carve((size_t)50016 * 96 * 2);  // max: conv3 A 50000x96 f16
  _Float16* Bt   = (_Float16*)carve((size_t)640 * 1024 * 2);  // max: fc2 W^T 512x1024 f16

  const int convC[3]   = {FFD, 2 * FFD, 4 * FFD};   // 33, 66, 132
  const int convKin[3] = {FFD, FFD, 2 * FFD};       // 33, 33, 66

  for (int br = 0; br < 2; ++br) {
    const float* x0    = (const float*)d_in[br * 3 + 0];
    const int*   ei    = (const int*)  d_in[br * 3 + 1];
    const int*   batch = (const int*)  d_in[br * 3 + 2];
    const int*   src   = ei;
    const int*   dst   = ei + EE;
    int base = 6 + br * 10;   // params: W1 b1 W2 b2 W3 b3 Wg1 bg1 Wg2 bg2
    const float* Wc[3] = {(const float*)d_in[base + 0], (const float*)d_in[base + 2],
                          (const float*)d_in[base + 4]};
    const float* bc[3] = {(const float*)d_in[base + 1], (const float*)d_in[base + 3],
                          (const float*)d_in[base + 5]};
    const float* Wg1 = (const float*)d_in[base + 6];
    const float* bg1 = (const float*)d_in[base + 7];
    const float* Wg2 = (const float*)d_in[base + 8];
    const float* bg2 = (const float*)d_in[base + 9];

    // symmetric degree normalization (dst side + self loop), shared across the 3 convs
    deg_init_kernel<<<cdiv(NN, 256), 256, 0, stream>>>(dinv, NN);
    deg_count_kernel<<<cdiv(EE, 256), 256, 0, stream>>>(dst, dinv, EE);
    deg_inv_kernel<<<cdiv(NN, 256), 256, 0, stream>>>(dinv, NN);

    const float* xin = x0;
    for (int l = 0; l < 3; ++l) {
      int C = convC[l], K = convKin[l];
      run_gemm(xin, Wc[l], nullptr, bufH, NN, C, K, C, 0, 0, Apad, Bt, stream); // h = x @ W
      agg_init_kernel<<<cdiv(NN * C, 256), 256, 0, stream>>>(bufH, dinv, bc[l], bufX, NN, C);
      edge_scatter_kernel<<<cdiv(EE * 32, 256), 256, 0, stream>>>(bufH, src, dst, dinv,
                                                                  bufX, EE, C);
      bool relu = (l < 2) || (br == 1);          // p2 has relu after conv3
      if (relu) relu_kernel<<<cdiv(NN * C, 256), 256, 0, stream>>>(bufX, NN * C);
      xin = bufX;
    }

    // global mean pool -> [512, 132]
    int C3 = convC[2];
    pool_zero_kernel<<<cdiv(BBG * C3, 256), 256, 0, stream>>>(pool, cnt, BBG, C3);
    pool_acc_kernel<<<cdiv(NN * 32, 256), 256, 0, stream>>>(bufX, batch, pool, cnt, NN, C3);
    pool_div_kernel<<<cdiv(BBG * C3, 256), 256, 0, stream>>>(pool, cnt, BBG, C3);

    // branch head: relu(pool @ Wg1 + bg1) @ Wg2 + bg2 -> xc[:, br*128 : br*128+128]
    run_gemm(pool, Wg1, bg1, g1, BBG, 1024, C3, 1024, 0, 1, Apad, Bt, stream);
    run_gemm(g1, Wg2, bg2, xc, BBG, 128, 1024, 256, br * 128, 0, Apad, Bt, stream);
  }

  // shared head
  const float* fc1W = (const float*)d_in[26]; const float* fc1b = (const float*)d_in[27];
  const float* fc2W = (const float*)d_in[28]; const float* fc2b = (const float*)d_in[29];
  const float* outW = (const float*)d_in[30]; const float* outb = (const float*)d_in[31];

  run_gemm(xc, fc1W, fc1b, f1, BBG, 1024, 256, 1024, 0, 1, Apad, Bt, stream);
  run_gemm(f1, fc2W, fc2b, f2, BBG, 512, 1024, 512, 0, 1, Apad, Bt, stream);
  out_layer_kernel<<<cdiv(BBG * 32, 256), 256, 0, stream>>>(f2, outW, outb,
                                                            (float*)d_out, BBG, 512);
}
